// LSTMGNN_45792941310334
// MI455X (gfx1250) — compile-verified, matrix operand order checked by reading, not run on hardware
//
#include <hip/hip_runtime.h>
#include <hip/hip_bf16.h>
#include <stdint.h>

// ---------------------------------------------------------------------------
// Problem constants (from reference)
// ---------------------------------------------------------------------------
#define B_   256
#define T_   200
#define A_   22
#define H_   256
#define N_   (B_ * A_)      // 5632 sequences
#define G3H  768            // 3*H gate width

typedef __attribute__((ext_vector_type(16))) __bf16 v16bf;
typedef __attribute__((ext_vector_type(8)))  __bf16 v8bf;
typedef __attribute__((ext_vector_type(8)))  float  v8f;

__device__ __forceinline__ unsigned short f2bf(float f) {
  union { float f; unsigned int u; } c; c.f = f;
  unsigned int u = c.u;
  u += 0x7FFFu + ((u >> 16) & 1u);          // round-to-nearest-even
  return (unsigned short)(u >> 16);
}

__device__ __forceinline__ float sigmoidf_(float x) {
  return 1.0f / (1.0f + __expf(-x));
}

// ---------------------------------------------------------------------------
// Generic bf16 GEMM: C[N,M] = A[N,K] @ W[M,K]^T (+ bias[M])
//   A: bf16 row-major with row stride lda (elements)
//   W: bf16 row-major [M,K]  (each output column m reads a contiguous W row)
//   C: fp32 row-major [N,M]
// Requires N%256==0, M%128==0, K%32==0 (true for all call sites).
// Block: 256 threads = 8 waves (4 in N x 2 in M); wave tile 64x64:
//   16 WMMAs per K-step vs 8 fragment loads -> 32 FLOP/B of L2 traffic.
// ---------------------------------------------------------------------------
__global__ __launch_bounds__(256)
void gemm_bf16_nt(const unsigned short* __restrict__ A, int lda,
                  const unsigned short* __restrict__ W,
                  const float* __restrict__ bias,
                  float* __restrict__ C, int M, int K) {
  const int lane = threadIdx.x & 31;
  const int wave = threadIdx.x >> 5;
  const int lo   = lane & 15;
  const int hi   = lane >> 4;            // 0|1 selects K sub-range per ISA layout
  const int koff = hi * 8;

  const long long rowBase = (long long)blockIdx.x * 256 + (wave & 3) * 64;
  const int       colBase = blockIdx.y * 128 + (wave >> 2) * 64;

  v8f acc[4][4];
#pragma unroll
  for (int i = 0; i < 4; ++i)
#pragma unroll
    for (int j = 0; j < 4; ++j) acc[i][j] = (v8f){};

  const unsigned short* ap[4];
  const unsigned short* bp[4];
#pragma unroll
  for (int i = 0; i < 4; ++i) {
    ap[i] = A + (rowBase + 16 * i + lo) * (long long)lda;
    bp[i] = W + (long long)(colBase + 16 * i + lo) * K;
  }

  for (int k = 0; k < K; k += 32) {
    const int k0 = k + koff;
    union Frag { v16bf v; v8bf h[2]; };
    Frag a[4], b[4];
#pragma unroll
    for (int i = 0; i < 4; ++i) {
      a[i].h[0] = *(const v8bf*)(ap[i] + k0);
      a[i].h[1] = *(const v8bf*)(ap[i] + k0 + 16);
    }
#pragma unroll
    for (int j = 0; j < 4; ++j) {
      b[j].h[0] = *(const v8bf*)(bp[j] + k0);
      b[j].h[1] = *(const v8bf*)(bp[j] + k0 + 16);
    }
#pragma unroll
    for (int i = 0; i < 4; ++i)
#pragma unroll
      for (int j = 0; j < 4; ++j)
        acc[i][j] = __builtin_amdgcn_wmma_f32_16x16x32_bf16(
            false, a[i].v, false, b[j].v, (short)0, acc[i][j], false, false);
  }

#pragma unroll
  for (int j = 0; j < 4; ++j) {
    const int col = colBase + 16 * j + lo;
    const float bi = bias ? bias[col] : 0.0f;
#pragma unroll
    for (int i = 0; i < 4; ++i) {
#pragma unroll
      for (int r = 0; r < 8; ++r) {
        const long long row = rowBase + 16 * i + r + hi * 8;   // C lane mapping
        C[row * M + col] = acc[i][j][r] + bi;
      }
    }
  }
}

// ---------------------------------------------------------------------------
// Prep: features[B,T,A,33] (fp32) -> Fbf[(b*A+a), t, 0..31] (bf16)
// ---------------------------------------------------------------------------
__global__ __launch_bounds__(256)
void prep_features(const float* __restrict__ feat, unsigned short* __restrict__ Fb) {
  long long idx = (long long)blockIdx.x * 256 + threadIdx.x;
  const long long total = (long long)N_ * T_ * 32;
  if (idx >= total) return;
  int k = (int)(idx & 31);
  long long r = idx >> 5;
  int t = (int)(r % T_); r /= T_;
  int a = (int)(r % A_);
  int b = (int)(r / A_);
  float v = feat[(((long long)b * T_ + t) * A_ + a) * 33 + k];
  Fb[idx] = f2bf(v);
}

// Wc[768,32] = wih1 @ fc_w (bf16); biasX1 = bih1 + wih1 @ fc_b
__global__ __launch_bounds__(256)
void prep_wc(const float* __restrict__ wih1, const float* __restrict__ fc_w,
             const float* __restrict__ fc_b, const float* __restrict__ bih1,
             float* __restrict__ biasX1, unsigned short* __restrict__ Wc) {
  int idx = blockIdx.x * 256 + threadIdx.x;
  if (idx >= G3H * 32) return;
  int m = idx >> 5, k = idx & 31;
  float s = 0.0f;
  for (int c = 0; c < H_; ++c) s += wih1[m * H_ + c] * fc_w[c * 32 + k];
  Wc[m * 32 + k] = f2bf(s);
  if (k == 0) {
    float bb = bih1[m];
    for (int c = 0; c < H_; ++c) bb += wih1[m * H_ + c] * fc_b[c];
    biasX1[m] = bb;
  }
}

__global__ __launch_bounds__(256)
void cvt_f32_bf16(const float* __restrict__ src, unsigned short* __restrict__ dst, int n) {
  int i = blockIdx.x * 256 + threadIdx.x;
  if (i < n) dst[i] = f2bf(src[i]);
}

__global__ __launch_bounds__(256)
void zero_b16(uint4* __restrict__ p, int n16) {   // zero n16 16-byte chunks
  int i = blockIdx.x * 256 + threadIdx.x;
  if (i < n16) p[i] = make_uint4(0u, 0u, 0u, 0u);
}

// ---------------------------------------------------------------------------
// GRU elementwise update: r=sig(Gx_r+Gh_r), z=sig(Gx_z+Gh_z),
// n=tanh(Gx_n + r*Gh_n); h=(1-z)n+z*h, frozen past seq length.
// ---------------------------------------------------------------------------
__global__ __launch_bounds__(256)
void gru_update(const float* __restrict__ Gx, const float* __restrict__ Gh,
                float* __restrict__ hf, unsigned short* __restrict__ hb,
                const int* __restrict__ seq_len, int t) {
  int idx = blockIdx.x * 256 + threadIdx.x;     // grid sized exactly N_*H_
  int n = idx >> 8, c = idx & 255;
  const float* gx = Gx + (long long)n * G3H;
  const float* gh = Gh + (long long)n * G3H;
  float r  = sigmoidf_(gx[c]        + gh[c]);
  float z  = sigmoidf_(gx[H_ + c]   + gh[H_ + c]);
  float nn = tanhf    (gx[2*H_ + c] + r * gh[2*H_ + c]);
  float h  = hf[idx];
  float hn = (1.0f - z) * nn + z * h;
  float out = (t < seq_len[n / A_]) ? hn : h;
  hf[idx] = out;
  hb[idx] = f2bf(out);
}

// ---------------------------------------------------------------------------
// Dense GAT attention over complete graph (A=22), heads in {1,4}, out=256.
// wx: fp32 [B, A, heads, 256]; writes relu(attn-mix + bias) as fp32 and/or bf16.
// One block per graph b.
// ---------------------------------------------------------------------------
__global__ __launch_bounds__(256)
void gat_attn(const float* __restrict__ wx,
              const float* __restrict__ a_src, const float* __restrict__ a_dst,
              const float* __restrict__ bias,
              float* __restrict__ outf, unsigned short* __restrict__ outb,
              int heads) {
  __shared__ float s_src[A_ * 4];
  __shared__ float s_dst[A_ * 4];
  __shared__ float s_attn[A_][A_ * 4 + 2];
  const int b = blockIdx.x;
  const float* w = wx + (long long)b * A_ * heads * 256;
  const int AH = A_ * heads;

  // per-node per-head source/dest scores
  for (int p = threadIdx.x; p < AH; p += 256) {
    int a = p / heads, h = p % heads;
    const float* wv = w + (a * heads + h) * 256;
    float ss = 0.0f, dd = 0.0f;
    for (int o = 0; o < 256; ++o) {
      float v = wv[o];
      ss += v * a_src[h * 256 + o];
      dd += v * a_dst[h * 256 + o];
    }
    s_src[p] = ss; s_dst[p] = dd;
  }
  __syncthreads();

  // softmax over source j for each (target i, head h)
  for (int p = threadIdx.x; p < AH; p += 256) {
    int i = p / heads, h = p % heads;
    float ev[A_];
    float mx = -1e30f;
    for (int j = 0; j < A_; ++j) {
      float x = s_dst[i * heads + h] + s_src[j * heads + h];
      x = (x > 0.0f) ? x : 0.2f * x;          // leaky_relu(0.2)
      ev[j] = x; mx = fmaxf(mx, x);
    }
    float sum = 0.0f;
    for (int j = 0; j < A_; ++j) { float e = __expf(ev[j] - mx); ev[j] = e; sum += e; }
    float inv = 1.0f / sum;
    for (int j = 0; j < A_; ++j) s_attn[i][j * heads + h] = ev[j] * inv;
  }
  __syncthreads();

  // o[i,h,:] = sum_j attn[i,j,h] * wx[j,h,:]  (+bias, relu)
  const int total = AH * 256;
  for (int p = threadIdx.x; p < total; p += 256) {
    int o = p & 255, ih = p >> 8;
    int i = ih / heads, h = ih % heads;
    float acc = 0.0f;
    for (int j = 0; j < A_; ++j)
      acc += s_attn[i][j * heads + h] * w[(j * heads + h) * 256 + o];
    acc += bias[h * 256 + o];
    acc = fmaxf(acc, 0.0f);
    long long oi = (long long)b * total + (long long)i * heads * 256 + h * 256 + o;
    if (outf) outf[oi] = acc;
    if (outb) outb[oi] = f2bf(acc);
  }
}

// mean-pool over A then dot with cls_w + cls_b -> logits[B]
__global__ __launch_bounds__(256)
void pool_cls(const float* __restrict__ xg, const float* __restrict__ cls_w,
              const float* __restrict__ cls_b, float* __restrict__ out) {
  __shared__ float red[256];
  const int b = blockIdx.x, tid = threadIdx.x;
  float m = 0.0f;
  for (int a = 0; a < A_; ++a) m += xg[((long long)b * A_ + a) * 256 + tid];
  m *= (1.0f / (float)A_);
  red[tid] = m * cls_w[tid];
  __syncthreads();
  for (int s = 128; s > 0; s >>= 1) {
    if (tid < s) red[tid] += red[tid + s];
    __syncthreads();
  }
  if (tid == 0) out[b] = red[0] + cls_b[0];
}

// ---------------------------------------------------------------------------
// Host-side orchestration
// ---------------------------------------------------------------------------
extern "C" void kernel_launch(void* const* d_in, const int* in_sizes, int n_in,
                              void* d_out, int out_size, void* d_ws, size_t ws_size,
                              hipStream_t stream) {
  const float* features = (const float*)d_in[0];
  const int*   seq_len  = (const int*)  d_in[1];
  const float* fc_w     = (const float*)d_in[2];
  const float* fc_b     = (const float*)d_in[3];
  const float* gru_wih  = (const float*)d_in[4];   // [2,768,256]
  const float* gru_whh  = (const float*)d_in[5];   // [2,768,256]
  const float* gru_bih  = (const float*)d_in[6];   // [2,768]
  const float* gru_bhh  = (const float*)d_in[7];   // [2,768]
  const float* gat_w0   = (const float*)d_in[8];
  const float* gat_as0  = (const float*)d_in[9];
  const float* gat_ad0  = (const float*)d_in[10];
  const float* gat_b0   = (const float*)d_in[11];
  const float* gat_w1   = (const float*)d_in[12];
  const float* gat_as1  = (const float*)d_in[13];
  const float* gat_ad1  = (const float*)d_in[14];
  const float* gat_b1   = (const float*)d_in[15];
  const float* gat_w2   = (const float*)d_in[16];
  const float* gat_as2  = (const float*)d_in[17];
  const float* gat_ad2  = (const float*)d_in[18];
  const float* gat_b2   = (const float*)d_in[19];
  const float* cls_w    = (const float*)d_in[20];
  const float* cls_b    = (const float*)d_in[21];

  // workspace carve-out (256B aligned)
  size_t off = 0;
  auto alloc = [&](size_t bytes) -> void* {
    void* p = (char*)d_ws + off;
    off += (bytes + 255) & ~(size_t)255;
    return p;
  };
  unsigned short* Fbf   = (unsigned short*)alloc((size_t)N_ * T_ * 32 * 2);   // 72 MB
  unsigned short* Wc    = (unsigned short*)alloc((size_t)G3H * 32 * 2);
  unsigned short* Whh1b = (unsigned short*)alloc((size_t)G3H * H_ * 2);
  unsigned short* Wih2b = (unsigned short*)alloc((size_t)G3H * H_ * 2);
  unsigned short* Whh2b = (unsigned short*)alloc((size_t)G3H * H_ * 2);
  unsigned short* Gw0b  = (unsigned short*)alloc((size_t)1024 * 256 * 2);
  unsigned short* Gw1b  = (unsigned short*)alloc((size_t)1024 * 1024 * 2);
  unsigned short* Gw2b  = (unsigned short*)alloc((size_t)256 * 1024 * 2);
  float*          biasX1= (float*)alloc((size_t)G3H * 4);
  float*          h1f   = (float*)alloc((size_t)N_ * H_ * 4);
  float*          h2f   = (float*)alloc((size_t)N_ * H_ * 4);
  unsigned short* h1b   = (unsigned short*)alloc((size_t)N_ * H_ * 2);
  unsigned short* h2b   = (unsigned short*)alloc((size_t)N_ * H_ * 2);
  float*          Gx    = (float*)alloc((size_t)N_ * G3H * 4);
  float*          Gh    = (float*)alloc((size_t)N_ * G3H * 4);
  float*          wxbuf = (float*)alloc((size_t)N_ * 1024 * 4);
  unsigned short* xg1b  = (unsigned short*)alloc((size_t)N_ * 1024 * 2);
  unsigned short* xg2b  = (unsigned short*)alloc((size_t)N_ * 1024 * 2);
  float*          xg3f  = (float*)alloc((size_t)N_ * 256 * 4);

  const dim3 blk(256);

  // ---- one-time (per call) weight prep ----
  {
    long long tot = (long long)N_ * T_ * 32;
    prep_features<<<dim3((unsigned)((tot + 255) / 256)), blk, 0, stream>>>(features, Fbf);
    prep_wc<<<dim3((G3H * 32 + 255) / 256), blk, 0, stream>>>(
        gru_wih, fc_w, fc_b, gru_bih, biasX1, Wc);
    cvt_f32_bf16<<<dim3((G3H * H_ + 255) / 256), blk, 0, stream>>>(gru_whh, Whh1b, G3H * H_);
    cvt_f32_bf16<<<dim3((G3H * H_ + 255) / 256), blk, 0, stream>>>(gru_wih + G3H * H_, Wih2b, G3H * H_);
    cvt_f32_bf16<<<dim3((G3H * H_ + 255) / 256), blk, 0, stream>>>(gru_whh + G3H * H_, Whh2b, G3H * H_);
    cvt_f32_bf16<<<dim3((1024 * 256 + 255) / 256), blk, 0, stream>>>(gat_w0, Gw0b, 1024 * 256);
    cvt_f32_bf16<<<dim3((1024 * 1024 + 255) / 256), blk, 0, stream>>>(gat_w1, Gw1b, 1024 * 1024);
    cvt_f32_bf16<<<dim3((256 * 1024 + 255) / 256), blk, 0, stream>>>(gat_w2, Gw2b, 256 * 1024);
    // zero h state (fp32 + bf16 shadows)
    int n16f = (N_ * H_ * 4) / 16, n16h = (N_ * H_ * 2) / 16;
    zero_b16<<<dim3((n16f + 255) / 256), blk, 0, stream>>>((uint4*)h1f, n16f);
    zero_b16<<<dim3((n16f + 255) / 256), blk, 0, stream>>>((uint4*)h2f, n16f);
    zero_b16<<<dim3((n16h + 255) / 256), blk, 0, stream>>>((uint4*)h1b, n16h);
    zero_b16<<<dim3((n16h + 255) / 256), blk, 0, stream>>>((uint4*)h2b, n16h);
  }

  // ---- GRU time loop: 2 layers interleaved per step ----
  const dim3 gGate(N_ / 256, G3H / 128);    // 22 x 6
  const dim3 gUpd(N_ * H_ / 256);           // 5632 blocks
  for (int t = 0; t < T_; ++t) {
    // layer 1: Gx = F_t @ Wc^T + biasX1 ; Gh = h1 @ whh1^T + bhh1
    gemm_bf16_nt<<<gGate, blk, 0, stream>>>(Fbf + (size_t)t * 32, T_ * 32, Wc,
                                            biasX1, Gx, G3H, 32);
    gemm_bf16_nt<<<gGate, blk, 0, stream>>>(h1b, H_, Whh1b, gru_bhh, Gh, G3H, H_);
    gru_update<<<gUpd, blk, 0, stream>>>(Gx, Gh, h1f, h1b, seq_len, t);
    // layer 2: Gx = h1 @ wih2^T + bih2 ; Gh = h2 @ whh2^T + bhh2
    gemm_bf16_nt<<<gGate, blk, 0, stream>>>(h1b, H_, Wih2b, gru_bih + G3H, Gx, G3H, H_);
    gemm_bf16_nt<<<gGate, blk, 0, stream>>>(h2b, H_, Whh2b, gru_bhh + G3H, Gh, G3H, H_);
    gru_update<<<gUpd, blk, 0, stream>>>(Gx, Gh, h2f, h2b, seq_len, t);
  }

  // ---- GAT stack (player_emb = h2, bf16 shadow already in h2b) ----
  gemm_bf16_nt<<<dim3(N_ / 256, 1024 / 128), blk, 0, stream>>>(
      h2b, H_, Gw0b, nullptr, wxbuf, 1024, 256);
  gat_attn<<<dim3(B_), blk, 0, stream>>>(wxbuf, gat_as0, gat_ad0, gat_b0,
                                         nullptr, xg1b, 4);
  gemm_bf16_nt<<<dim3(N_ / 256, 1024 / 128), blk, 0, stream>>>(
      xg1b, 1024, Gw1b, nullptr, wxbuf, 1024, 1024);
  gat_attn<<<dim3(B_), blk, 0, stream>>>(wxbuf, gat_as1, gat_ad1, gat_b1,
                                         nullptr, xg2b, 4);
  gemm_bf16_nt<<<dim3(N_ / 256, 256 / 128), blk, 0, stream>>>(
      xg2b, 1024, Gw2b, nullptr, wxbuf, 256, 1024);
  gat_attn<<<dim3(B_), blk, 0, stream>>>(wxbuf, gat_as2, gat_ad2, gat_b2,
                                         xg3f, nullptr, 1);

  // ---- global mean pool + classifier ----
  pool_cls<<<dim3(B_), blk, 0, stream>>>(xg3f, cls_w, cls_b, (float*)d_out);
}
